// FeedForward_32023276159510
// MI455X (gfx1250) — compile-verified, matrix operand order checked by reading, not run on hardware
//
#include <hip/hip_runtime.h>
#include <hip/hip_bf16.h>

typedef __attribute__((ext_vector_type(16))) _Float16 v16h;
typedef __attribute__((ext_vector_type(8)))  _Float16 v8h;
typedef __attribute__((ext_vector_type(4)))  _Float16 v4h;
typedef __attribute__((ext_vector_type(8)))  float    v8f;

#define LDT 136  // LDS row stride in halves (128 data + 8 pad); 272B, 16B-aligned

// ---------------------------------------------------------------------------
// 16x32 f16 operand fragment from an LDS tile (row stride LDT halves).
// Lanes 0-15: row=lane, K={0..7,16..23}; lanes 16-31: row=lane-16, K={8..15,24..31}.
__device__ __forceinline__ v16h lds_frag(const _Float16* tile, int row0, int k0, int lane) {
  const int r  = row0 + (lane & 15);
  const int ko = k0 + ((lane >> 4) << 3);
  const _Float16* p = tile + r * LDT + ko;
  v8h lo = *(const v8h*)(p);
  v8h hi = *(const v8h*)(p + 16);
  v16h o;
#pragma unroll
  for (int i = 0; i < 8; ++i) { o[i] = lo[i]; o[i + 8] = hi[i]; }
  return o;
}

// ---------------------------------------------------------------------------
// Async copy of a 128-row x 128-half f16 tile (32 KB) global -> LDS.
// ASYNCcnt-tracked, no VGPR staging. 256 threads x 8 chunks of 16B.
__device__ __forceinline__ void async_tile(const _Float16* gbase, int strideHalves,
                                           _Float16* lbuf, int tid) {
#pragma unroll
  for (int i = 0; i < 8; ++i) {
    const int c   = tid + i * 256;          // chunk 0..2047
    const int row = c >> 4;                 // 0..127
    const int seg = c & 15;                 // 16B segment within 256B row
    const unsigned voff = (unsigned)(row * strideHalves * 2 + seg * 16);
    const unsigned lds  = (unsigned)(uintptr_t)(lbuf + row * LDT + seg * 8);
    asm volatile("global_load_async_to_lds_b128 %0, %1, %2 offset:0"
                 :: "v"(lds), "v"(voff), "s"(gbase)
                 : "memory");
  }
}

__device__ __forceinline__ void wait_async0() {
  asm volatile("s_wait_asynccnt 0x0" ::: "memory");
}

// ---------------------------------------------------------------------------
// Dequant 16 packed Q4_0 ints (one 128-K block's worth for this thread):
// MSB nibbles -> K chunk [0,64), LSB nibbles -> K chunk [64,128). Each packed
// int is fetched from HBM exactly once.
__device__ __forceinline__ void dequant_store2(const int4* q, float scm, float scl,
                                               _Float16* dst /* row base + seg*16 */) {
#pragma unroll
  for (int i = 0; i < 4; ++i) {
    int4 p = q[i];
    v4h m = { (_Float16)((p.x >> 4) * scm), (_Float16)((p.y >> 4) * scm),
              (_Float16)((p.z >> 4) * scm), (_Float16)((p.w >> 4) * scm) };
    v4h l = { (_Float16)(((( p.x & 15) ^ 8) - 8) * scl),
              (_Float16)(((( p.y & 15) ^ 8) - 8) * scl),
              (_Float16)(((( p.z & 15) ^ 8) - 8) * scl),
              (_Float16)(((( p.w & 15) ^ 8) - 8) * scl) };
    ((v4h*)dst)[i]      = m;   // K [0,64) of this 128-block
    ((v4h*)(dst + 64))[i] = l; // K [64,128)
  }
}

#define WMMA_F16(A, B, C) \
  __builtin_amdgcn_wmma_f32_16x16x32_f16(false, (A), false, (B), (short)0, (C), false, false)

// ---------------------------------------------------------------------------
// Kernel 0: one-time x f32 -> f16.
__global__ __launch_bounds__(256) void cvt_x_kernel(const float* __restrict__ x,
                                                    _Float16* __restrict__ x16) {
  const int i = (blockIdx.x * 256 + threadIdx.x) * 8;
  float4 a = *(const float4*)(x + i);
  float4 b = *(const float4*)(x + i + 4);
  v4h o0 = { (_Float16)a.x, (_Float16)a.y, (_Float16)a.z, (_Float16)a.w };
  v4h o1 = { (_Float16)b.x, (_Float16)b.y, (_Float16)b.z, (_Float16)b.w };
  *(v4h*)(x16 + i)     = o0;
  *(v4h*)(x16 + i + 4) = o1;
}

// ---------------------------------------------------------------------------
// Kernel 1: h = silu(x@W1^T) * (x@W3^T), f16 out.
// Block 128(M) x 64(N), 8 waves in 4x2, per-wave 32x32 via 2x2 WMMA tiles.
// K-step 128: A async double-buffered in LDS, weights register-prefetched and
// dequantized once per packed int (both nibbles).
__global__ __launch_bounds__(256) void ffn_gateup_kernel(
    const _Float16* __restrict__ x16,
    const int* __restrict__ w1, const float* __restrict__ s1,
    const int* __restrict__ w3, const float* __restrict__ s3,
    _Float16* __restrict__ h, int dim, int hidden) {
  __shared__ _Float16 XS[2][128 * LDT];
  __shared__ _Float16 W1S[64 * LDT];
  __shared__ _Float16 W3S[64 * LDT];

  const int tid  = threadIdx.x;
  const int lane = tid & 31;
  const int wave = tid >> 5;
  const int wm = wave >> 1;
  const int wn = wave & 1;
  const int mBase = blockIdx.y * 128;
  const int nBase = blockIdx.x * 64;

  const v8f zero = {0.f, 0.f, 0.f, 0.f, 0.f, 0.f, 0.f, 0.f};
  v8f accg[2][2], accu[2][2];
#pragma unroll
  for (int i = 0; i < 2; ++i)
#pragma unroll
    for (int j = 0; j < 2; ++j) { accg[i][j] = zero; accu[i][j] = zero; }

  // Weight prefetch: this thread covers (wrow, 16 packed ints) of a 64x128-K block.
  const int wrow = tid >> 2, wseg = tid & 3;
  int4 pw1[4], pw3[4];
  float2 sc1, sc3;
  auto loadW = [&](int kb) {
    const int f0 = (nBase + wrow) * dim + kb;          // kb is 128-aligned
    sc1 = *(const float2*)(s1 + (f0 >> 6));            // {msb scale, lsb scale}
    sc3 = *(const float2*)(s3 + (f0 >> 6));
    const int4* p1 = (const int4*)(w1 + ((f0 >> 7) << 6) + wseg * 16);
    const int4* p3 = (const int4*)(w3 + ((f0 >> 7) << 6) + wseg * 16);
#pragma unroll
    for (int i = 0; i < 4; ++i) { pw1[i] = p1[i]; pw3[i] = p3[i]; }
  };

  auto stage = [&]() {  // dequant current W regs + make A tile visible
    dequant_store2(pw1, sc1.x, sc1.y, W1S + wrow * LDT + wseg * 16);
    dequant_store2(pw3, sc3.x, sc3.y, W3S + wrow * LDT + wseg * 16);
    wait_async0();
    __syncthreads();
  };

  auto mma = [&](const _Float16* Xc) {
#pragma unroll
    for (int ks = 0; ks < 128; ks += 32) {
      v16h a0 = lds_frag(Xc,  wm * 32 +  0, ks, lane);
      v16h a1 = lds_frag(Xc,  wm * 32 + 16, ks, lane);
      v16h g0 = lds_frag(W1S, wn * 32 +  0, ks, lane);
      v16h g1 = lds_frag(W1S, wn * 32 + 16, ks, lane);
      v16h u0 = lds_frag(W3S, wn * 32 +  0, ks, lane);
      v16h u1 = lds_frag(W3S, wn * 32 + 16, ks, lane);
      accg[0][0] = WMMA_F16(a0, g0, accg[0][0]);
      accg[0][1] = WMMA_F16(a0, g1, accg[0][1]);
      accg[1][0] = WMMA_F16(a1, g0, accg[1][0]);
      accg[1][1] = WMMA_F16(a1, g1, accg[1][1]);
      accu[0][0] = WMMA_F16(a0, u0, accu[0][0]);
      accu[0][1] = WMMA_F16(a0, u1, accu[0][1]);
      accu[1][0] = WMMA_F16(a1, u0, accu[1][0]);
      accu[1][1] = WMMA_F16(a1, u1, accu[1][1]);
    }
  };

  const _Float16* arow = x16 + (size_t)mBase * dim;
  const int niter = dim / 128;

  loadW(0);
  async_tile(arow, dim, XS[0], tid);

  for (int it = 0; it < niter - 1; ++it) {   // steady state: branch-free body
    stage();
    loadW((it + 1) * 128);                          // overlap with WMMA burst
    async_tile(arow + (it + 1) * 128, dim, XS[(it + 1) & 1], tid);
    mma(XS[it & 1]);
    __syncthreads();                                 // protect W LDS reuse
  }
  stage();                                           // peeled last iteration
  mma(XS[(niter - 1) & 1]);

  // SwiGLU epilogue; C/D layout: VGPR r, lanes 0-15 -> M=r, lanes 16-31 -> M=8+r.
#pragma unroll
  for (int mt = 0; mt < 2; ++mt)
#pragma unroll
    for (int nt = 0; nt < 2; ++nt) {
      const int col  = nBase + wn * 32 + nt * 16 + (lane & 15);
      const int rowb = mBase + wm * 32 + mt * 16 + ((lane >> 4) << 3);
      v8f g = accg[mt][nt], u = accu[mt][nt];
#pragma unroll
      for (int r = 0; r < 8; ++r) {
        float a = g[r];
        float val = (a / (1.f + __expf(-a))) * u[r];
        h[(size_t)(rowb + r) * hidden + col] = (_Float16)val;
      }
    }
}

// ---------------------------------------------------------------------------
// Kernel 2: out = h @ W2^T, f32 out. Same pipeline, single accumulator set.
__global__ __launch_bounds__(256) void ffn_down_kernel(
    const _Float16* __restrict__ h,
    const int* __restrict__ w2, const float* __restrict__ s2,
    float* __restrict__ out, int dim, int hidden) {
  __shared__ _Float16 HS[2][128 * LDT];
  __shared__ _Float16 W2S[64 * LDT];

  const int tid  = threadIdx.x;
  const int lane = tid & 31;
  const int wave = tid >> 5;
  const int wm = wave >> 1;
  const int wn = wave & 1;
  const int mBase = blockIdx.y * 128;
  const int nBase = blockIdx.x * 64;

  const v8f zero = {0.f, 0.f, 0.f, 0.f, 0.f, 0.f, 0.f, 0.f};
  v8f acc[2][2];
#pragma unroll
  for (int i = 0; i < 2; ++i)
#pragma unroll
    for (int j = 0; j < 2; ++j) acc[i][j] = zero;

  const int wrow = tid >> 2, wseg = tid & 3;
  int4 pw2[4];
  float2 sc2;
  auto loadW = [&](int kb) {
    const int f0 = (nBase + wrow) * hidden + kb;
    sc2 = *(const float2*)(s2 + (f0 >> 6));
    const int4* p2 = (const int4*)(w2 + ((f0 >> 7) << 6) + wseg * 16);
#pragma unroll
    for (int i = 0; i < 4; ++i) pw2[i] = p2[i];
  };

  auto stage = [&]() {
    dequant_store2(pw2, sc2.x, sc2.y, W2S + wrow * LDT + wseg * 16);
    wait_async0();
    __syncthreads();
  };

  auto mma = [&](const _Float16* Hc) {
#pragma unroll
    for (int ks = 0; ks < 128; ks += 32) {
      v16h a0 = lds_frag(Hc,  wm * 32 +  0, ks, lane);
      v16h a1 = lds_frag(Hc,  wm * 32 + 16, ks, lane);
      v16h b0 = lds_frag(W2S, wn * 32 +  0, ks, lane);
      v16h b1 = lds_frag(W2S, wn * 32 + 16, ks, lane);
      acc[0][0] = WMMA_F16(a0, b0, acc[0][0]);
      acc[0][1] = WMMA_F16(a0, b1, acc[0][1]);
      acc[1][0] = WMMA_F16(a1, b0, acc[1][0]);
      acc[1][1] = WMMA_F16(a1, b1, acc[1][1]);
    }
  };

  const _Float16* arow = h + (size_t)mBase * hidden;
  const int niter = hidden / 128;

  loadW(0);
  async_tile(arow, hidden, HS[0], tid);

  for (int it = 0; it < niter - 1; ++it) {
    stage();
    loadW((it + 1) * 128);
    async_tile(arow + (it + 1) * 128, hidden, HS[(it + 1) & 1], tid);
    mma(HS[it & 1]);
    __syncthreads();
  }
  stage();
  mma(HS[(niter - 1) & 1]);

#pragma unroll
  for (int mt = 0; mt < 2; ++mt)
#pragma unroll
    for (int nt = 0; nt < 2; ++nt) {
      const int col  = nBase + wn * 32 + nt * 16 + (lane & 15);
      const int rowb = mBase + wm * 32 + mt * 16 + ((lane >> 4) << 3);
      v8f c = acc[mt][nt];
#pragma unroll
      for (int r = 0; r < 8; ++r)
        out[(size_t)(rowb + r) * dim + col] = c[r];
    }
}

// ---------------------------------------------------------------------------
extern "C" void kernel_launch(void* const* d_in, const int* in_sizes, int n_in,
                              void* d_out, int out_size, void* d_ws, size_t ws_size,
                              hipStream_t stream) {
  const float* x  = (const float*)d_in[0];
  const int*   w1 = (const int*)d_in[1];
  const float* s1 = (const float*)d_in[2];
  const int*   w2 = (const int*)d_in[3];
  const float* s2 = (const float*)d_in[4];
  const int*   w3 = (const int*)d_in[5];
  const float* s3 = (const float*)d_in[6];
  float* out = (float*)d_out;

  const int dim    = 4096;
  const int hidden = (in_sizes[1] * 2) / dim;   // 11008
  const int M      = in_sizes[0] / dim;         // 4096 tokens

  _Float16* x16  = (_Float16*)d_ws;             // M x dim f16     (~34 MB)
  _Float16* hbuf = x16 + (size_t)M * dim;       // M x hidden f16  (~90 MB)

  dim3 blk(256);
  cvt_x_kernel<<<dim3(((size_t)M * dim) / 2048), blk, 0, stream>>>(x, x16);
  ffn_gateup_kernel<<<dim3(hidden / 64, M / 128), blk, 0, stream>>>(
      x16, w1, s1, w3, s3, hbuf, dim, hidden);
  ffn_down_kernel<<<dim3(dim / 64, M / 128), blk, 0, stream>>>(
      hbuf, w2, s2, out, dim, hidden);
}
